// BayesianDecoder_62405874810902
// MI455X (gfx1250) — matrix operand order repaired and checked by reading, not compiled, not run on hardware
//
#include <hip/hip_runtime.h>
#include <hip/hip_bf16.h>

typedef __attribute__((ext_vector_type(16))) _Float16 v16h;
typedef __attribute__((ext_vector_type(8)))  float    v8f;

#define DROP_KEEP_THRESH 0xE6666666u   /* 0.9 * 2^32 */
#define DROP_SCALE       (1.0f / 0.9f)

__device__ __forceinline__ unsigned h32(unsigned x) {
  x ^= x >> 16; x *= 0x7feb352du;
  x ^= x >> 15; x *= 0x846ca68bu;
  x ^= x >> 16; return x;
}
__device__ __forceinline__ float u01(unsigned x) {
  return ((float)(x >> 8) + 0.5f) * (1.0f / 16777216.0f);
}
__device__ __forceinline__ float gaussf(unsigned s) {
  unsigned a = h32(s * 2u + 0x9E3779B9u);
  unsigned b = h32(s ^ 0x94d049bbu);
  float r = __builtin_sqrtf(-2.0f * __logf(u01(a)));
  return r * __cosf(6.28318530718f * u01(b));
}
__device__ __forceinline__ float softplusf(float r) {
  return (r > 15.0f) ? r : log1pf(__expf(r));
}

// ---------------------------------------------------------------------------
// Stages 1-4: unpool (DEG=4 gather-weighted-sum) + BN(batch) + ReLU + dropout.
// One wave handles 16 fine nodes x 16 batches. Lane l works on node (l&15);
// lanes 0-15 pack Y, lanes 16-31 pack Y^2 into one f16 B operand (32x16).
// Two v_wmma_f32_16x16x32_f16 with constant half-ones A matrices give
// per-node sum and sum-of-squares in f32 (all D rows equal the column sum).
// Intermediates are node-major [N][16] so gathers are contiguous 64B rows.
// ---------------------------------------------------------------------------
template <int STAGE, bool COARSE_BATCH_MAJOR>
__global__ __launch_bounds__(256) void unpool_bn_relu_drop(
    const float* __restrict__ xin, const int* __restrict__ ii,
    const float* __restrict__ mu, const float* __restrict__ rho,
    const float* __restrict__ gg, const float* __restrict__ bb,
    float* __restrict__ out, int n_coarse)
{
  const int lane = threadIdx.x & 31;
  const int wave = threadIdx.x >> 5;
  const int j    = (blockIdx.x * 8 + wave) * 16 + (lane & 15);
  const bool hiHalf = lane >= 16;

  float yv[16];
#pragma unroll
  for (int b = 0; b < 16; ++b) yv[b] = 0.0f;

#pragma unroll
  for (int d = 0; d < 4; ++d) {
    const int e = 4 * j + d;
    __builtin_prefetch(&mu[e + 512], 0, 1);
    __builtin_prefetch(&rho[e + 512], 0, 1);
    const int   idx = ii[e];
    const float sg  = softplusf(rho[e]);
    const float w   = mu[e] + sg * gaussf(((unsigned)STAGE << 27) ^ (unsigned)e);
    if (COARSE_BATCH_MAJOR) {
#pragma unroll
      for (int b = 0; b < 16; ++b)
        yv[b] += xin[(size_t)b * (size_t)n_coarse + (size_t)idx] * w;
    } else {
      const float4* xr = (const float4*)(xin + (size_t)idx * 16);
      float4 r0 = xr[0], r1 = xr[1], r2 = xr[2], r3 = xr[3];
      yv[0]  += r0.x * w; yv[1]  += r0.y * w; yv[2]  += r0.z * w; yv[3]  += r0.w * w;
      yv[4]  += r1.x * w; yv[5]  += r1.y * w; yv[6]  += r1.z * w; yv[7]  += r1.w * w;
      yv[8]  += r2.x * w; yv[9]  += r2.y * w; yv[10] += r2.z * w; yv[11] += r2.w * w;
      yv[12] += r3.x * w; yv[13] += r3.y * w; yv[14] += r3.z * w; yv[15] += r3.w * w;
    }
  }

  // Pack B operand: lanes 0-15 carry Y (K=0..15), lanes 16-31 carry Y^2
  // (K=16..31).  A1 selects K<16 (-> sum Y), A2 selects K>=16 (-> sum Y^2).
  v16h Bm = {};
#pragma unroll
  for (int b = 0; b < 16; ++b) {
    float v = yv[b];
    Bm[b] = (_Float16)(hiHalf ? v * v : v);
  }
  v16h A1 = {}, A2 = {};
#pragma unroll
  for (int i = 0; i < 16; ++i) { A1[i] = (_Float16)(hiHalf ? 0.0f : 1.0f); A2[i] = (_Float16)(hiHalf ? 1.0f : 0.0f); }

  v8f S = {}, Q = {};
  S = __builtin_amdgcn_wmma_f32_16x16x32_f16(false, A1, false, Bm, (short)0, S, false, false);
  Q = __builtin_amdgcn_wmma_f32_16x16x32_f16(false, A2, false, Bm, (short)0, Q, false, false);

  const float mean = S[0] * 0.0625f;
  const float var  = Q[0] * 0.0625f - mean * mean;
  const float rs   = rsqrtf(var + 1e-5f);
  const float gj   = gg[j];
  const float bj   = bb[j];

#pragma unroll
  for (int b = 0; b < 16; ++b) {
    float t = (yv[b] - mean) * rs * gj + bj;
    t = fmaxf(t, 0.0f);
    unsigned u = h32(((unsigned)(j * 16 + b)) * 0x85EBCA6Bu ^
                     ((unsigned)STAGE * 0xC2B2AE35u) ^ 0xD1B54A35u);
    yv[b] = (u < DROP_KEEP_THRESH) ? t * DROP_SCALE : 0.0f;
  }

  // Each lane-half stores its 8 batches of the node row (node-major [N][16]).
  float a0, a1, a2, a3, a4, a5, a6, a7;
  if (hiHalf) { a0=yv[8];  a1=yv[9];  a2=yv[10]; a3=yv[11];
                a4=yv[12]; a5=yv[13]; a6=yv[14]; a7=yv[15]; }
  else        { a0=yv[0];  a1=yv[1];  a2=yv[2];  a3=yv[3];
                a4=yv[4];  a5=yv[5];  a6=yv[6];  a7=yv[7]; }
  float4* op = (float4*)(out + (size_t)j * 16 + (hiHalf ? 8 : 0));
  op[0] = make_float4(a0, a1, a2, a3);
  op[1] = make_float4(a4, a5, a6, a7);
}

// ---------------------------------------------------------------------------
// Stage 5: unpool with Cout=3, no BN/ReLU/dropout; writes batch-major output
// [B=16][nf][3].  One thread per fine node; 48 accumulators.
// ---------------------------------------------------------------------------
__global__ __launch_bounds__(256) void unpool_out(
    const float* __restrict__ xin, const int* __restrict__ ii,
    const float* __restrict__ mu, const float* __restrict__ rho,
    float* __restrict__ out, int nf)
{
  const int j = blockIdx.x * 256 + threadIdx.x;
  float acc[16][3];
#pragma unroll
  for (int b = 0; b < 16; ++b)
#pragma unroll
    for (int o = 0; o < 3; ++o) acc[b][o] = 0.0f;

#pragma unroll
  for (int d = 0; d < 4; ++d) {
    const int e = 4 * j + d;
    __builtin_prefetch(&mu[(size_t)(e + 256) * 3], 0, 1);
    __builtin_prefetch(&rho[(size_t)(e + 256) * 3], 0, 1);
    const int idx = ii[e];
    float w[3];
#pragma unroll
    for (int o = 0; o < 3; ++o) {
      const size_t we = (size_t)e * 3 + o;
      const float sg = softplusf(rho[we]);
      w[o] = mu[we] + sg * gaussf(0xA0000000u ^ (unsigned)we);
    }
    const float4* xr = (const float4*)(xin + (size_t)idx * 16);
    float4 r0 = xr[0], r1 = xr[1], r2 = xr[2], r3 = xr[3];
    const float xv[16] = { r0.x, r0.y, r0.z, r0.w, r1.x, r1.y, r1.z, r1.w,
                           r2.x, r2.y, r2.z, r2.w, r3.x, r3.y, r3.z, r3.w };
#pragma unroll
    for (int b = 0; b < 16; ++b)
#pragma unroll
      for (int o = 0; o < 3; ++o) acc[b][o] += xv[b] * w[o];
  }

  const size_t plane = (size_t)nf * 3;
#pragma unroll
  for (int b = 0; b < 16; ++b) {
    const size_t base = (size_t)b * plane + (size_t)j * 3;
    out[base + 0] = acc[b][0];
    out[base + 1] = acc[b][1];
    out[base + 2] = acc[b][2];
  }
}

// ---------------------------------------------------------------------------
// KL reduction: sum over all variational params of 0.5(s^2+m^2) - ln s - 0.5
// ---------------------------------------------------------------------------
__global__ __launch_bounds__(256) void kl_zero(float* p) { *p = 0.0f; }

__global__ __launch_bounds__(256) void kl_kernel(
    const float* __restrict__ mu, const float* __restrict__ rho,
    int n, float* __restrict__ kl_out)
{
  __shared__ float sm[8];
  const int i = blockIdx.x * 256 + threadIdx.x;
  float v = 0.0f;
  if (i < n) {
    const float m = mu[i];
    const float s = softplusf(rho[i]);
    v = 0.5f * (s * s + m * m) - __logf(s) - 0.5f;
  }
#pragma unroll
  for (int off = 16; off > 0; off >>= 1) v += __shfl_down(v, off, 32);
  if ((threadIdx.x & 31) == 0) sm[threadIdx.x >> 5] = v;
  __syncthreads();
  if (threadIdx.x == 0) {
    float t = 0.0f;
#pragma unroll
    for (int w = 0; w < 8; ++w) t += sm[w];
    atomicAdd(kl_out, t);
  }
}

extern "C" void kernel_launch(void* const* d_in, const int* in_sizes, int n_in,
                              void* d_out, int out_size, void* d_ws, size_t ws_size,
                              hipStream_t stream) {
  (void)in_sizes; (void)n_in; (void)ws_size;
  // setup_inputs() order: x, then per stage i: io, ii, mu, rho, (g, b for i<5)
  const float* x    = (const float*)d_in[0];
  const int*   ii1  = (const int*)  d_in[2];
  const float* mu1  = (const float*)d_in[3];
  const float* rho1 = (const float*)d_in[4];
  const float* g1   = (const float*)d_in[5];
  const float* b1   = (const float*)d_in[6];
  const int*   ii2  = (const int*)  d_in[8];
  const float* mu2  = (const float*)d_in[9];
  const float* rho2 = (const float*)d_in[10];
  const float* g2   = (const float*)d_in[11];
  const float* b2   = (const float*)d_in[12];
  const int*   ii3  = (const int*)  d_in[14];
  const float* mu3  = (const float*)d_in[15];
  const float* rho3 = (const float*)d_in[16];
  const float* g3   = (const float*)d_in[17];
  const float* b3   = (const float*)d_in[18];
  const int*   ii4  = (const int*)  d_in[20];
  const float* mu4  = (const float*)d_in[21];
  const float* rho4 = (const float*)d_in[22];
  const float* g4   = (const float*)d_in[23];
  const float* b4   = (const float*)d_in[24];
  const int*   ii5  = (const int*)  d_in[26];
  const float* mu5  = (const float*)d_in[27];
  const float* rho5 = (const float*)d_in[28];

  // node-major [N][16] intermediates in workspace (fully rewritten each call)
  float* h1 = (float*)d_ws;
  float* h2 = h1 + (size_t)2048  * 16;
  float* h3 = h2 + (size_t)8192  * 16;
  float* h4 = h3 + (size_t)32768 * 16;

  float* out = (float*)d_out;
  float* kl  = out + ((size_t)out_size - 1);

  unpool_bn_relu_drop<1, true ><<<2048   / 128, 256, 0, stream>>>(x,  ii1, mu1, rho1, g1, b1, h1, 512);
  unpool_bn_relu_drop<2, false><<<8192   / 128, 256, 0, stream>>>(h1, ii2, mu2, rho2, g2, b2, h2, 2048);
  unpool_bn_relu_drop<3, false><<<32768  / 128, 256, 0, stream>>>(h2, ii3, mu3, rho3, g3, b3, h3, 8192);
  unpool_bn_relu_drop<4, false><<<131072 / 128, 256, 0, stream>>>(h3, ii4, mu4, rho4, g4, b4, h4, 32768);
  unpool_out<<<524288 / 256, 256, 0, stream>>>(h4, ii5, mu5, rho5, out, 524288);

  kl_zero<<<1, 1, 0, stream>>>(kl);
  kl_kernel<<<(8192    + 255) / 256, 256, 0, stream>>>(mu1, rho1, 8192,    kl);
  kl_kernel<<<(32768   + 255) / 256, 256, 0, stream>>>(mu2, rho2, 32768,   kl);
  kl_kernel<<<(131072  + 255) / 256, 256, 0, stream>>>(mu3, rho3, 131072,  kl);
  kl_kernel<<<(524288  + 255) / 256, 256, 0, stream>>>(mu4, rho4, 524288,  kl);
  kl_kernel<<<(6291456 + 255) / 256, 256, 0, stream>>>(mu5, rho5, 6291456, kl);
}
